// MultiHeadAttention_88931592831578
// MI455X (gfx1250) — matrix operand order repaired and checked by reading, not compiled, not run on hardware
//
#include <hip/hip_runtime.h>
#include <hip/hip_bf16.h>
#include <stdint.h>

// ---------------------------------------------------------------------------
// MHA block for MI455X (gfx1250).
//   pre-pass: x -> bf16, w_qkv -> bf16 transposed, w_out -> bf16 transposed
//   k_qkv_gemm : bf16 WMMA GEMM, async-to-LDS double-buffered tiles,
//                writes Q/K head-major and V d-major (pre-transposed)
//   k_rope     : in-place RoPE on Q/K
//   k_attn     : causal flash attention, async K/V tiles, WMMA scores+PV
//   k_oproj    : bf16 WMMA GEMM -> fp32 output
// All matrix math: v_wmma_f32_16x16x32_bf16 (fp32 WMMA is only 16x16x4 on
// CDNA5 -> 8x less K per issue, so bf16 is the throughput precision).
// Tile staging: global_load_async_to_lds_b128 (ASYNCcnt) so tiles stream in
// while WMMAs consume the previous buffer.
// ---------------------------------------------------------------------------

typedef __attribute__((ext_vector_type(16))) __bf16 v16bf;
typedef __attribute__((ext_vector_type(8)))  float  v8f;

#define BB    4
#define SS    2048
#define DD    2048
#define HH    16
#define HD    128
#define N_QKV 6144

__device__ __forceinline__ unsigned short f2bf(float f) {
    unsigned int u = __float_as_uint(f);
    u += 0x7FFFu + ((u >> 16) & 1u);          // round-to-nearest-even
    return (unsigned short)(u >> 16);
}
__device__ __forceinline__ float bf2f(unsigned short h) {
    return __uint_as_float(((unsigned int)h) << 16);
}

union Frag {                 // 16 bf16 = 8 VGPRs, loaded as 2x b128
    v16bf v;
    uint4 q[2];
};

__device__ __forceinline__ v8f zero8() {
    v8f z; for (int i = 0; i < 8; ++i) z[i] = 0.f; return z;
}

// A-matrix fragment (16x32 bf16, MxK): lane holds row (lane&15);
// K pairs at element offsets kb..kb+7 and kb+16..kb+23, kb = (lane>>4)*8.
__device__ __forceinline__ Frag load_fragA(const unsigned short* rowp, int half) {
    Frag f; const int kb = half * 8;
    f.q[0] = *(const uint4*)(rowp + kb);
    f.q[1] = *(const uint4*)(rowp + kb + 16);
    return f;
}
// B-matrix fragment (32x16 bf16, KxN): lane holds col (lane&15);
// K at element offsets kb..kb+7 and kb+16..kb+23, kb = (lane>>4)*8 (matches A).
__device__ __forceinline__ Frag load_fragB(const unsigned short* colp, int half) {
    Frag f; const int kb = half * 8;
    f.q[0] = *(const uint4*)(colp + kb);
    f.q[1] = *(const uint4*)(colp + kb + 16);
    return f;
}

__device__ __forceinline__ v8f wmma_bf16(const Frag& a, const Frag& b, v8f c) {
    return __builtin_amdgcn_wmma_f32_16x16x32_bf16(
        false, a.v, false, b.v, (short)0, c, false, false);
}

// Async global->LDS 16-byte copy (ASYNCcnt).  The LDS-address VGPR takes the
// low 32 bits of the flat address (HW truncates flat LDS addrs to [31:0]).
__device__ __forceinline__ void async_copy16(void* lds, const void* gptr) {
    const unsigned lo = (unsigned)(size_t)lds;
    asm volatile("global_load_async_to_lds_b128 %0, %1, off"
                 :: "v"(lo), "v"(gptr) : "memory");
}
__device__ __forceinline__ void wait_async0() {
    asm volatile("s_wait_asynccnt 0x0" ::: "memory");
}

// ---------------------------------------------------------------------------
// Pre-pass kernels: one-time fp32 -> bf16 conversion (and weight transpose)
// so the GEMM hot loops stage pure bf16 with zero VALU work.
// ---------------------------------------------------------------------------
__global__ __launch_bounds__(256)
void k_cvt(const float* __restrict__ src, unsigned short* __restrict__ dst, size_t n) {
    const size_t i = ((size_t)blockIdx.x * 256 + threadIdx.x) * 4;
    if (i >= n) return;
    const float4 f = *(const float4*)(src + i);
    uint2 u;
    u.x = (unsigned)f2bf(f.x) | ((unsigned)f2bf(f.y) << 16);
    u.y = (unsigned)f2bf(f.z) | ((unsigned)f2bf(f.w) << 16);
    *(uint2*)(dst + i) = u;
}

// src [K][N] fp32 row-major -> dst [N][K] bf16 (transposed)
__global__ __launch_bounds__(256)
void k_cvt_t(const float* __restrict__ src, unsigned short* __restrict__ dst,
             int K, int N) {
    const int chunks = (K * N) >> 2;
    const int gid = blockIdx.x * 256 + threadIdx.x;
    if (gid >= chunks) return;
    const int k  = gid / (N >> 2);
    const int nc = (gid - k * (N >> 2)) << 2;
    const float4 f = *(const float4*)(src + (size_t)k * N + nc);
    dst[(size_t)(nc + 0) * K + k] = f2bf(f.x);
    dst[(size_t)(nc + 1) * K + k] = f2bf(f.y);
    dst[(size_t)(nc + 2) * K + k] = f2bf(f.z);
    dst[(size_t)(nc + 3) * K + k] = f2bf(f.w);
}

// ---------------------------------------------------------------------------
// Kernel: QKV GEMM.  C[M=B*S, N=3D] = xb * wqkvT^T, all-bf16 tiles staged by
// async copies with double buffering.  Q/K stored head-major [B][H][S][HD];
// V stored d-major [B][H][HD][S] so attention can async-copy V^T directly.
// ---------------------------------------------------------------------------
__global__ __launch_bounds__(256)
void k_qkv_gemm(const unsigned short* __restrict__ xb,
                const unsigned short* __restrict__ wT,   // [N_QKV][DD] bf16
                unsigned short* __restrict__ qb, unsigned short* __restrict__ kb,
                unsigned short* __restrict__ vT) {
    __shared__ __align__(16) unsigned short sA[2][128 * 32];
    __shared__ __align__(16) unsigned short sB[2][128 * 32];

    const int tid  = threadIdx.x;
    const int lane = tid & 31;
    const int wave = tid >> 5;
    const int half = lane >> 4;
    const int ln   = lane & 15;
    const int n0   = blockIdx.x * 128;
    const int m0   = blockIdx.y * 128;
    const int wm   = wave & 1;
    const int wn   = wave >> 1;

    v8f acc[4][2];
    for (int i = 0; i < 4; ++i) for (int j = 0; j < 2; ++j) acc[i][j] = zero8();

    // 128x32 bf16 tile = 512 x 16B chunks; 2 chunks per thread per operand.
    auto issue = [&](int k0, int buf) {
        for (int i = 0; i < 2; ++i) {
            const int c   = tid + i * 256;
            const int row = c >> 2, kc = (c & 3) << 3;
            async_copy16(&sA[buf][row * 32 + kc],
                         xb + (size_t)(m0 + row) * DD + k0 + kc);
            async_copy16(&sB[buf][row * 32 + kc],
                         wT + (size_t)(n0 + row) * DD + k0 + kc);
        }
    };

    issue(0, 0);
    for (int k0 = 0; k0 < DD; k0 += 32) {
        const int buf = (k0 >> 5) & 1;
        wait_async0();                         // tile[buf] landed (this wave)
        __syncthreads();                       // ... and every other wave's
        if (k0 + 32 < DD) issue(k0 + 32, buf ^ 1);   // prefetch next tile

        Frag a[4], b[2];
        for (int mt = 0; mt < 4; ++mt)
            a[mt] = load_fragA(&sA[buf][(wm * 64 + mt * 16 + ln) * 32], half);
        for (int nt = 0; nt < 2; ++nt)
            b[nt] = load_fragB(&sB[buf][(wn * 32 + nt * 16 + ln) * 32], half);
        for (int mt = 0; mt < 4; ++mt)
            for (int nt = 0; nt < 2; ++nt)
                acc[mt][nt] = wmma_bf16(a[mt], b[nt], acc[mt][nt]);
    }

    // C frag: M = r + 8*half, N = ln.  Scatter bf16 into q/k head-major and
    // v d-major (pre-transposed for the attention kernel's B operand).
    for (int mt = 0; mt < 4; ++mt)
        for (int nt = 0; nt < 2; ++nt)
            for (int r = 0; r < 8; ++r) {
                const int m = m0 + wm * 64 + mt * 16 + r + 8 * half;
                const int n = n0 + wn * 32 + nt * 16 + ln;
                const int which = n >> 11;               // 0=q 1=k 2=v
                const int rem   = n & 2047;
                const int h = rem >> 7, d = rem & 127;
                const int bi = m >> 11, s = m & 2047;
                const unsigned short val = f2bf(acc[mt][nt][r]);
                if (which == 2)
                    vT[((size_t)(bi * HH + h) * HD + d) * SS + s] = val;
                else {
                    unsigned short* dst = (which == 0) ? qb : kb;
                    dst[((size_t)(bi * HH + h) * SS + s) * HD + d] = val;
                }
            }
}

// ---------------------------------------------------------------------------
// Kernel: RoPE in place on Q/K (bf16, head-major).  Bandwidth-trivial.
// ---------------------------------------------------------------------------
__global__ __launch_bounds__(256)
void k_rope(unsigned short* __restrict__ qb, unsigned short* __restrict__ kb,
            const float* __restrict__ cosT, const float* __restrict__ sinT) {
    const size_t rows = (size_t)BB * HH * SS;
    const size_t per  = rows * 64;
    size_t g = (size_t)blockIdx.x * blockDim.x + threadIdx.x;
    if (g >= 2 * per) return;
    unsigned short* buf = qb;
    if (g >= per) { buf = kb; g -= per; }
    const size_t row = g >> 6;
    const int d = (int)(g & 63);
    const int s = (int)(row & (SS - 1));
    unsigned short* p = buf + row * HD;
    const float x1 = bf2f(p[d]), x2 = bf2f(p[d + 64]);
    const float c = cosT[s * 64 + d], sn = sinT[s * 64 + d];
    p[d]      = f2bf(x1 * c - x2 * sn);
    p[d + 64] = f2bf(x1 * sn + x2 * c);
}

// ---------------------------------------------------------------------------
// Kernel: causal flash attention.  128 Q rows per block (8 waves x 16 rows).
// Q fragments VGPR-resident; K tile [kcol][d] and V^T tile [d][krow] arrive
// via double-buffered async copies (V is already d-major in global memory).
// Per 32-key step per wave: 8 WMMA scores, shuffle-based online softmax,
// P via per-wave LDS (C-layout -> A-layout), 8 WMMA for PV.
// ---------------------------------------------------------------------------
__global__ __launch_bounds__(256)
void k_attn(const unsigned short* __restrict__ qg, const unsigned short* __restrict__ kg,
            const unsigned short* __restrict__ vTg, unsigned short* __restrict__ og) {
    __shared__ __align__(16) unsigned short sK[2][32 * 128];   // [kcol][d]
    __shared__ __align__(16) unsigned short sV[2][128 * 32];   // [d][krow]
    __shared__ __align__(16) unsigned short sP[8][16 * 32];    // per-wave P

    const int tid = threadIdx.x, lane = tid & 31, wave = tid >> 5;
    const int half = lane >> 4, ln = lane & 15;
    const int bh = blockIdx.y;
    const int q0 = blockIdx.x * 128;

    const unsigned short* qbase  = qg  + (size_t)bh * SS * HD;
    const unsigned short* kbase  = kg  + (size_t)bh * SS * HD;
    const unsigned short* vTbase = vTg + (size_t)bh * HD * SS;
    unsigned short*       obase  = og  + (size_t)bh * SS * HD;

    // Q fragments resident for the whole loop (4 K-steps of 32 over hd=128)
    const int qrowA = q0 + wave * 16 + ln;
    Frag qf[4];
    for (int ks = 0; ks < 4; ++ks)
        qf[ks] = load_fragA(qbase + (size_t)qrowA * HD + ks * 32, half);

    v8f o[8];
    for (int f = 0; f < 8; ++f) o[f] = zero8();
    float mprev[8], lsum[8];
    for (int r = 0; r < 8; ++r) { mprev[r] = -1e30f; lsum[r] = 0.f; }

    const int myrow0  = q0 + wave * 16;
    const int j_end_w = (myrow0 + 15) / 32 + 1;      // wave's causal limit
    const int j_end_b = (q0 + 127) / 32 + 1;         // block max
    const float scale = 0.08838834764831845f;        // 1/sqrt(128)

    // K tile: 512 chunks (32 rows x 16 chunks); V^T: 512 (128 rows x 4).
    auto issue = [&](int j, int buf) {
        for (int i = 0; i < 2; ++i) {
            const int c  = tid + i * 256;
            const int kr = c >> 4, dc = (c & 15) << 3;
            async_copy16(&sK[buf][kr * 128 + dc],
                         kbase + (size_t)(j * 32 + kr) * HD + dc);
            const int dr = c >> 2, kc = (c & 3) << 3;
            async_copy16(&sV[buf][dr * 32 + kc],
                         vTbase + (size_t)dr * SS + j * 32 + kc);
        }
    };

    issue(0, 0);
    for (int j = 0; j < j_end_b; ++j) {
        const int buf = j & 1;
        wait_async0();
        __syncthreads();
        if (j + 1 < j_end_b) issue(j + 1, buf ^ 1);
        if (j >= j_end_w) continue;                   // wave-uniform skip

        // scores: S = Q (16x128) x K^T (128x32)
        v8f sc[2];
        for (int nt = 0; nt < 2; ++nt) {
            v8f a = zero8();
            for (int ks = 0; ks < 4; ++ks) {
                Frag kf = load_fragB(&sK[buf][(nt * 16 + ln) * 128 + ks * 32], half);
                a = wmma_bf16(qf[ks], kf, a);
            }
            sc[nt] = a;
        }

        // online softmax: per-row stats in 8 scalars/lane; rows of a half are
        // spread across its 16 lanes -> xor-butterfly reductions.
        for (int r = 0; r < 8; ++r) {
            const int rowg = myrow0 + r + 8 * half;
            float s0 = sc[0][r] * scale;
            float s1 = sc[1][r] * scale;
            const int c0 = j * 32 + ln;
            if (c0      > rowg) s0 = -1e30f;          // causal mask
            if (c0 + 16 > rowg) s1 = -1e30f;
            float mx = fmaxf(s0, s1);
            mx = fmaxf(mx, __shfl_xor(mx, 1));
            mx = fmaxf(mx, __shfl_xor(mx, 2));
            mx = fmaxf(mx, __shfl_xor(mx, 4));
            mx = fmaxf(mx, __shfl_xor(mx, 8));
            const float mnew = fmaxf(mprev[r], mx);
            const float corr = __expf(mprev[r] - mnew);
            const float p0 = __expf(s0 - mnew);
            const float p1 = __expf(s1 - mnew);
            float rs = p0 + p1;
            rs += __shfl_xor(rs, 1);
            rs += __shfl_xor(rs, 2);
            rs += __shfl_xor(rs, 4);
            rs += __shfl_xor(rs, 8);
            lsum[r] = lsum[r] * corr + rs;
            mprev[r] = mnew;
            for (int f = 0; f < 8; ++f) o[f][r] *= corr;
            sP[wave][(r + 8 * half) * 32 + ln]      = f2bf(p0);
            sP[wave][(r + 8 * half) * 32 + 16 + ln] = f2bf(p1);
        }

        // PV: O += P (16x32) x V (32x128).  LDS is in-order per wave, so the
        // P scratch round-trip needs no barrier.
        Frag pf = load_fragA(&sP[wave][ln * 32], half);
        for (int f = 0; f < 8; ++f) {
            Frag vf = load_fragB(&sV[buf][(f * 16 + ln) * 32], half);
            o[f] = wmma_bf16(pf, vf, o[f]);
        }
    }

    for (int f = 0; f < 8; ++f)
        for (int r = 0; r < 8; ++r) {
            const int rowg = myrow0 + r + 8 * half;
            obase[(size_t)rowg * HD + f * 16 + ln] = f2bf(o[f][r] / lsum[r]);
        }
}

// ---------------------------------------------------------------------------
// Kernel: output projection.  out[M=B*S, N=D] = attn (bf16 head-major gather)
// * w_outT^T (bf16).  Async double-buffered tiles; fp32 output.
// ---------------------------------------------------------------------------
__global__ __launch_bounds__(256)
void k_oproj(const unsigned short* __restrict__ attn,
             const unsigned short* __restrict__ wT,     // [DD][DD] bf16 (N-major)
             float* __restrict__ out) {
    __shared__ __align__(16) unsigned short sA[2][128 * 32];
    __shared__ __align__(16) unsigned short sB[2][128 * 32];

    const int tid  = threadIdx.x;
    const int lane = tid & 31;
    const int wave = tid >> 5;
    const int half = lane >> 4;
    const int ln   = lane & 15;
    const int n0   = blockIdx.x * 128;
    const int m0   = blockIdx.y * 128;
    const int wm   = wave & 1;
    const int wn   = wave >> 1;

    v8f acc[4][2];
    for (int i = 0; i < 4; ++i) for (int j = 0; j < 2; ++j) acc[i][j] = zero8();

    auto issue = [&](int k0, int buf) {
        for (int i = 0; i < 2; ++i) {
            const int c   = tid + i * 256;
            const int row = c >> 2, kc = (c & 3) << 3;
            // A: gather bf16 from [B][H][S][HD]; a 16B chunk never crosses a
            // head boundary (kc is a multiple of 8, HD = 128).
            const int m = m0 + row, bi = m >> 11, s = m & 2047;
            const int k = k0 + kc,  h = k >> 7,  d = k & 127;
            async_copy16(&sA[buf][row * 32 + kc],
                         attn + ((size_t)(bi * HH + h) * SS + s) * HD + d);
            async_copy16(&sB[buf][row * 32 + kc],
                         wT + (size_t)(n0 + row) * DD + k0 + kc);
        }
    };

    issue(0, 0);
    for (int k0 = 0; k0 < DD; k0 += 32) {
        const int buf = (k0 >> 5) & 1;
        wait_async0();
        __syncthreads();
        if (k0 + 32 < DD) issue(k0 + 32, buf ^ 1);

        Frag a[4], b[2];
        for (int mt = 0; mt < 4; ++mt)
            a[mt] = load_fragA(&sA[buf][(wm * 64 + mt * 16 + ln) * 32], half);
        for (int nt = 0; nt < 2; ++nt)
            b[nt] = load_fragB(&sB[buf][(wn * 32 + nt * 16 + ln) * 32], half);
        for (int mt = 0; mt < 4; ++mt)
            for (int nt = 0; nt < 2; ++nt)
                acc[mt][nt] = wmma_bf16(a[mt], b[nt], acc[mt][nt]);
    }

    for (int mt = 0; mt < 4; ++mt)
        for (int nt = 0; nt < 2; ++nt)
            for (int r = 0; r < 8; ++r) {
                const int m = m0 + wm * 64 + mt * 16 + r + 8 * half;
                const int n = n0 + wn * 32 + nt * 16 + ln;
                out[(size_t)m * DD + n] = acc[mt][nt][r];
            }
}

// ---------------------------------------------------------------------------
extern "C" void kernel_launch(void* const* d_in, const int* in_sizes, int n_in,
                              void* d_out, int out_size, void* d_ws, size_t ws_size,
                              hipStream_t stream) {
    const float* x    = (const float*)d_in[0];
    const float* cosT = (const float*)d_in[1];
    const float* sinT = (const float*)d_in[2];
    const float* wqkv = (const float*)d_in[3];
    const float* wout = (const float*)d_in[4];

    // workspace layout (bf16 buffers), total 192 MB:
    //   qb/kb [B][H][S][HD] 32MB each | vT [B][H][HD][S] 32MB | ab 32MB
    //   xb [B*S][D] 32MB | wqT [3D][D] 24MB | woT [D][D] 8MB
    const size_t nTok = (size_t)BB * HH * SS * HD;        // 16M elems
    unsigned short* qb  = (unsigned short*)d_ws;
    unsigned short* kb  = qb + nTok;
    unsigned short* vT  = kb + nTok;
    unsigned short* ab  = vT + nTok;
    unsigned short* xb  = ab + nTok;
    unsigned short* wqT = xb + (size_t)BB * SS * DD;
    unsigned short* woT = wqT + (size_t)N_QKV * DD;

    // one-time conversions / transposes
    const size_t nX = (size_t)BB * SS * DD;
    k_cvt<<<(unsigned)(nX / 4 / 256), 256, 0, stream>>>(x, xb, nX);
    k_cvt_t<<<(unsigned)((size_t)DD * N_QKV / 4 / 256), 256, 0, stream>>>(wqkv, wqT, DD, N_QKV);
    k_cvt_t<<<(unsigned)((size_t)DD * DD / 4 / 256), 256, 0, stream>>>(wout, woT, DD, DD);

    dim3 g1(N_QKV / 128, (BB * SS) / 128);                // 48 x 64
    k_qkv_gemm<<<g1, 256, 0, stream>>>(xb, wqT, qb, kb, vT);

    const size_t ropeThreads = 2ull * BB * HH * SS * 64;
    k_rope<<<(unsigned)((ropeThreads + 255) / 256), 256, 0, stream>>>(qb, kb, cosT, sinT);

    dim3 g3(SS / 128, BB * HH);                           // 16 x 64
    k_attn<<<g3, 256, 0, stream>>>(qb, kb, vT, ab);

    dim3 g4(DD / 128, (BB * SS) / 128);                   // 16 x 64
    k_oproj<<<g4, 256, 0, stream>>>(ab, woT, (float*)d_out);
}
